// ScannedLSTM_48515950576424
// MI455X (gfx1250) — compile-verified, matrix operand order checked by reading, not run on hardware
//
#include <hip/hip_runtime.h>

// ---------------------------------------------------------------------------
// Scanned LSTM with done-reset on MI455X (gfx1250).
//   Per step: z = [xt | masked h] @ Wc + b   (Wc = [Wi;Wh], 2048x4096 bf16,
//   L2-resident), WMMA bf16 16x16x32 with f32 accumulation, gates in f32.
//   4 waves per workgroup split K four ways (16 tiles each) to quarter the
//   sequential per-step critical path and quadruple device wave count; LDS
//   reduction; ping-pong (unroll-2) register pipeline.
// ---------------------------------------------------------------------------

typedef __attribute__((ext_vector_type(16))) __bf16 v16bf;
typedef __attribute__((ext_vector_type(8)))  float  v8f;

#define T_STEPS 512
#define B_DIM   128
#define H_DIM   1024
#define K_DIM   2048            // [x | h]
#define Z_DIM   4096            // 4 gates * H
#define KT      (K_DIM / 32)    // 64 K-tiles of 32
#define KT_Q    (KT / 4)        // 16 K-tiles per wave
#define NTH     (H_DIM / 16)    // 64 column tiles per gate

__device__ __forceinline__ float sigm(float x) {
  return __frcp_rn(1.0f + __expf(-x));          // saturates cleanly
}
__device__ __forceinline__ float tanh_fast(float x) {
  return 1.0f - 2.0f * __frcp_rn(1.0f + __expf(2.0f * x));
}

struct F16 { float v[16]; };

// One lane's worth of a 16x32 A-tile source: two contiguous runs of 8 f32.
__device__ __forceinline__ F16 loadA16(const float* __restrict__ s, int kb) {
  F16 r;
  const float4 a = *(const float4*)(s + kb);
  const float4 b = *(const float4*)(s + kb + 4);
  const float4 c = *(const float4*)(s + kb + 16);
  const float4 d = *(const float4*)(s + kb + 20);
  r.v[0]=a.x;  r.v[1]=a.y;  r.v[2]=a.z;  r.v[3]=a.w;
  r.v[4]=b.x;  r.v[5]=b.y;  r.v[6]=b.z;  r.v[7]=b.w;
  r.v[8]=c.x;  r.v[9]=c.y;  r.v[10]=c.z; r.v[11]=c.w;
  r.v[12]=d.x; r.v[13]=d.y; r.v[14]=d.z; r.v[15]=d.w;
  return r;
}

__device__ __forceinline__ v16bf cvtA(const F16& f) {
  v16bf a;
#pragma unroll
  for (int e = 0; e < 16; ++e) a[e] = (__bf16)f.v[e];   // -> v_cvt_pk_bf16_f32 x8
  return a;
}

// ---------------------------------------------------------------------------
// Repack [Wi;Wh] (f32, row-major 2048x4096) into bf16 tiles laid out exactly
// as the wave32 WMMA B-operand (ISA 7.12.2): tile (nt_z, kt) = 512 bf16 as
// [lane][e], lane -> (N = lane&15, K-half = lane>>4). Each lane's 16 values
// are contiguous => two global_load_b128 per tile in the step kernel.
// ---------------------------------------------------------------------------
__global__ void repack_weights(const float* __restrict__ Wi,
                               const float* __restrict__ Wh,
                               __bf16* __restrict__ Wp) {
  const int idx    = blockIdx.x * blockDim.x + threadIdx.x;  // 0 .. 2048*4096-1
  const int within = idx & 511;
  const int tile   = idx >> 9;
  const int kt     = tile & (KT - 1);
  const int nt_z   = tile / KT;            // 0..255 (z column tile)
  const int lane   = within >> 4;
  const int e      = within & 15;
  const int n      = nt_z * 16 + (lane & 15);
  const int kb     = (lane >> 4) * 8;
  const int j      = e >> 1;
  const int bit    = e & 1;
  const int kloc   = (j < 4) ? (kb + 2 * j + bit) : (16 + kb + 2 * (j - 4) + bit);
  const int k      = kt * 32 + kloc;
  const float v = (k < H_DIM) ? Wi[k * Z_DIM + n] : Wh[(k - H_DIM) * Z_DIM + n];
  Wp[idx] = (__bf16)v;
}

__global__ void init_state(const float* __restrict__ c0, const float* __restrict__ h0,
                           float* __restrict__ c, float* __restrict__ h) {
  const int i = blockIdx.x * blockDim.x + threadIdx.x;
  c[i] = c0[i];
  h[i] = h0[i];
}

// ---------------------------------------------------------------------------
// One LSTM timestep.  Grid: 512 workgroups x 4 waves (128 threads).
//   wid -> (mt: 16-row M tile, nt: 16-col h tile); all 4 gates per wave.
//   Wave w covers K quarter w*512..w*512+511: waves 0,1 = x half,
//   waves 2,3 = h half (row-masked at the reduction).
//   Waves 1..3 spill partials through LDS; wave 0 sums + gate epilogue.
// ---------------------------------------------------------------------------
__global__ __launch_bounds__(128)
void lstm_step(const float* __restrict__ x,            // [T,B,H]
               const unsigned char* __restrict__ dones,// [T,B] bool
               const float* __restrict__ bias,         // [4H]
               const __bf16* __restrict__ Wp,          // packed [Wi;Wh]
               float* __restrict__ c_buf,              // [B,H] in/out
               const float* __restrict__ h_prev,       // [B,H]
               float* __restrict__ h_next,             // [B,H]
               float* __restrict__ ys,                 // [T,B,H] (d_out)
               int t) {
  __shared__ float lds[3 * 4 * 256];      // partials from waves 1..3 (12 KB)

  const int tid  = threadIdx.x;
  const int wave = tid >> 5;              // 0..3 : K quarter
  const int lane = tid & 31;
  const int wid  = blockIdx.x;            // 0..511
  const int mt   = wid >> 6;              // 0..7
  const int nt   = wid & 63;              // 0..63

  const int lm   = lane & 15;
  const int kb   = (lane >> 4) * 8;
  const int m    = mt * 16 + lm;
  const int ncol = nt * 16 + lm;

  // Source rows: waves 0,1 read xt; waves 2,3 read h_prev.  Column offset
  // within the half: 512 floats for the odd wave of each pair.
  const int koff = (wave & 1) * (KT_Q * 32);
  const float* srow = ((wave < 2) ? (x + (t * B_DIM + m) * H_DIM)
                                  : (h_prev + m * H_DIM)) + koff;

  // Accumulators per gate; bias (depends only on N = lane&15) into wave 0.
  v8f acc[4];
#pragma unroll
  for (int g = 0; g < 4; ++g) {
    const float bg = (wave == 0) ? bias[g * H_DIM + ncol] : 0.0f;
#pragma unroll
    for (int r = 0; r < 8; ++r) acc[g][r] = bg;
  }

  // Per-gate packed-B pointers for this wave's K quarter.
  const __bf16* bp[4];
#pragma unroll
  for (int g = 0; g < 4; ++g)
    bp[g] = Wp + (size_t)(((g * NTH + nt) * KT) + wave * KT_Q) * 512 + lane * 16;

  // --- ping-pong pipelined K loop (2 tiles/iter, no in-loop guards) -------
  F16   a0 = loadA16(srow, kb);
  F16   a1;
  v16bf b0[4], b1[4];
#pragma unroll
  for (int g = 0; g < 4; ++g) b0[g] = *(const v16bf*)bp[g];

  for (int i = 0; i < KT_Q - 2; i += 2) {
    // prefetch tile i+1 -> buf1
    a1 = loadA16(srow + (i + 1) * 32, kb);
#pragma unroll
    for (int g = 0; g < 4; ++g)
      b1[g] = *(const v16bf*)(bp[g] + (size_t)(i + 1) * 512);
    {
      const v16bf A = cvtA(a0);
#pragma unroll
      for (int g = 0; g < 4; ++g)
        acc[g] = __builtin_amdgcn_wmma_f32_16x16x32_bf16(
                   false, A, false, b0[g], (short)0, acc[g], false, false);
    }
    // prefetch tile i+2 -> buf0
    a0 = loadA16(srow + (i + 2) * 32, kb);
#pragma unroll
    for (int g = 0; g < 4; ++g)
      b0[g] = *(const v16bf*)(bp[g] + (size_t)(i + 2) * 512);
    {
      const v16bf A = cvtA(a1);
#pragma unroll
      for (int g = 0; g < 4; ++g)
        acc[g] = __builtin_amdgcn_wmma_f32_16x16x32_bf16(
                   false, A, false, b1[g], (short)0, acc[g], false, false);
    }
  }
  // peeled tail: tiles KT_Q-2 (in buf0) and KT_Q-1
  a1 = loadA16(srow + (KT_Q - 1) * 32, kb);
#pragma unroll
  for (int g = 0; g < 4; ++g)
    b1[g] = *(const v16bf*)(bp[g] + (size_t)(KT_Q - 1) * 512);
  {
    const v16bf A = cvtA(a0);
#pragma unroll
    for (int g = 0; g < 4; ++g)
      acc[g] = __builtin_amdgcn_wmma_f32_16x16x32_bf16(
                 false, A, false, b0[g], (short)0, acc[g], false, false);
  }
  {
    const v16bf A = cvtA(a1);
#pragma unroll
    for (int g = 0; g < 4; ++g)
      acc[g] = __builtin_amdgcn_wmma_f32_16x16x32_bf16(
                 false, A, false, b1[g], (short)0, acc[g], false, false);
  }

  // --- cross-wave reduction through LDS -----------------------------------
  // done-mask zeroes ROWS of the h-half partial product (waves 2,3 only).
  if (wave != 0) {
    const bool hhalf = (wave >= 2);       // wave-uniform
    float rowm[8];
#pragma unroll
    for (int r = 0; r < 8; ++r) {
      const int M = (lane < 16) ? r : (8 + r);
      rowm[r] = (hhalf && dones[t * B_DIM + mt * 16 + M]) ? 0.0f : 1.0f;
    }
    const int slot = wave - 1;            // 0..2
#pragma unroll
    for (int g = 0; g < 4; ++g) {
      float* p = &lds[(slot * 4 + g) * 256 + lane * 8];
      *(float4*)(p + 0) = make_float4(acc[g][0] * rowm[0], acc[g][1] * rowm[1],
                                      acc[g][2] * rowm[2], acc[g][3] * rowm[3]);
      *(float4*)(p + 4) = make_float4(acc[g][4] * rowm[4], acc[g][5] * rowm[5],
                                      acc[g][6] * rowm[6], acc[g][7] * rowm[7]);
    }
  }
  __syncthreads();

  if (wave == 0) {
#pragma unroll
    for (int g = 0; g < 4; ++g) {
#pragma unroll
      for (int s = 0; s < 3; ++s) {
        const float4 p0 = *(const float4*)&lds[(s * 4 + g) * 256 + lane * 8];
        const float4 p1 = *(const float4*)&lds[(s * 4 + g) * 256 + lane * 8 + 4];
        acc[g][0] += p0.x; acc[g][1] += p0.y; acc[g][2] += p0.z; acc[g][3] += p0.w;
        acc[g][4] += p1.x; acc[g][5] += p1.y; acc[g][6] += p1.z; acc[g][7] += p1.w;
      }
    }

    // Epilogue: C/D layout -> VGPR r: M = r (+8 for lanes 16..31), N = lane&15.
#pragma unroll
    for (int r = 0; r < 8; ++r) {
      const int   M    = (lane < 16) ? r : (8 + r);
      const int   mr   = mt * 16 + M;
      const int   ci   = mr * H_DIM + ncol;
      const float iv   = sigm(acc[0][r]);
      const float fv   = sigm(acc[1][r]);
      const float gv   = tanh_fast(acc[2][r]);
      const float ov   = sigm(acc[3][r]);
      const float cold = dones[t * B_DIM + mr] ? 0.0f : c_buf[ci];
      const float cnew = fv * cold + iv * gv;
      const float hnew = ov * tanh_fast(cnew);
      c_buf[ci]  = cnew;
      h_next[ci] = hnew;
      ys[(t * B_DIM + mr) * H_DIM + ncol] = hnew;
    }
  }
}

__global__ void copy_final(const float* __restrict__ c, const float* __restrict__ h,
                           float* __restrict__ out) {
  const int i = blockIdx.x * blockDim.x + threadIdx.x;   // 0..B*H-1
  out[i] = c[i];
  out[B_DIM * H_DIM + i] = h[i];
}

extern "C" void kernel_launch(void* const* d_in, const int* in_sizes, int n_in,
                              void* d_out, int out_size, void* d_ws, size_t ws_size,
                              hipStream_t stream) {
  const float*         x     = (const float*)d_in[0];
  const unsigned char* dones = (const unsigned char*)d_in[1];
  const float*         Wi    = (const float*)d_in[2];
  const float*         Wh    = (const float*)d_in[3];
  const float*         b     = (const float*)d_in[4];
  const float*         c0    = (const float*)d_in[5];
  const float*         h0    = (const float*)d_in[6];
  float*               out   = (float*)d_out;

  // Workspace: 16 MB packed bf16 weights + c + h ping-pong (~17.5 MB total).
  char*   ws    = (char*)d_ws;
  __bf16* Wp    = (__bf16*)ws;
  float*  c_buf = (float*)(ws + (size_t)K_DIM * Z_DIM * sizeof(__bf16));
  float*  hb0   = c_buf + B_DIM * H_DIM;
  float*  hb1   = hb0 + B_DIM * H_DIM;

  repack_weights<<<(K_DIM * Z_DIM) / 256, 256, 0, stream>>>(Wi, Wh, Wp);
  init_state<<<(B_DIM * H_DIM) / 256, 256, 0, stream>>>(c0, h0, c_buf, hb0);

  for (int t = 0; t < T_STEPS; ++t) {
    const float* hp = (t & 1) ? hb1 : hb0;
    float*       hn = (t & 1) ? hb0 : hb1;
    lstm_step<<<512, 128, 0, stream>>>(x, dones, b, Wp, c_buf, hp, hn, out, t);
  }

  const float* hf = ((T_STEPS - 1) & 1) ? hb0 : hb1;  // buffer written at t=T-1
  copy_final<<<(B_DIM * H_DIM) / 256, 256, 0, stream>>>(
      c_buf, hf, out + (size_t)T_STEPS * B_DIM * H_DIM);
}
